// DualAttention_63161789055235
// MI455X (gfx1250) — compile-verified, matrix operand order checked by reading, not run on hardware
//
#include <hip/hip_runtime.h>
#include <hip/hip_bf16.h>

typedef __bf16 bf16;
typedef __attribute__((ext_vector_type(8)))  bf16  v8bf;
typedef __attribute__((ext_vector_type(16))) bf16  v16bf;
typedef __attribute__((ext_vector_type(8)))  float v8f;

#define BS 8
#define LL 1024
#define DD 512
#define BPAD 40   // padded row stride (elements) for transposed LDS B tile

// ---------- WMMA fragment helper (bf16, 16x16x32) ----------
// Lane L: m = L&15, ksel = L>>4.  Per-lane elements:
//   t<8 : K = k0 + ksel*8 + t        t>=8 : K = k0 + 16 + ksel*8 + (t-8)
// caller passes p = row_ptr + k0 + ksel*8  (K-contiguous source, global or LDS)
__device__ inline v16bf load_frag16(const bf16* p) {
    v8bf lo = *(const v8bf*)(p);
    v8bf hi = *(const v8bf*)(p + 16);
    v16bf r;
#pragma unroll
    for (int t = 0; t < 8; ++t) { r[t] = lo[t]; r[t + 8] = hi[t]; }
    return r;
}

// ---------- 1) bf16 staging: Exb, Eyb, Eyw_b = (Ey * w3) ----------
__global__ void prep_kernel(const float* __restrict__ Ex, const float* __restrict__ Ey,
                            const float* __restrict__ W,
                            bf16* __restrict__ Exb, bf16* __restrict__ Eyb,
                            bf16* __restrict__ Eywb, long n) {
    for (long idx = blockIdx.x * 256L + threadIdx.x; idx < n; idx += (long)gridDim.x * 256L) {
        int c = (int)(idx & (DD - 1));
        float ex = Ex[idx], ey = Ey[idx];
        Exb[idx]  = (bf16)ex;
        Eyb[idx]  = (bf16)ey;
        Eywb[idx] = (bf16)(ey * W[2 * DD + c]);
    }
}

// ---------- 2) alpha = Ex@w1, beta = Ey@w2 ----------
__global__ void alphabeta_kernel(const float* __restrict__ Ex, const float* __restrict__ Ey,
                                 const float* __restrict__ W,
                                 float* __restrict__ alpha, float* __restrict__ beta) {
    int r = blockIdx.x * blockDim.x + threadIdx.x;   // 0 .. BS*LL-1
    if (r >= BS * LL) return;
    const float* ex = Ex + (size_t)r * DD;
    const float* ey = Ey + (size_t)r * DD;
    float sa = 0.f, sb = 0.f;
    for (int d = 0; d < DD; ++d) { sa += ex[d] * W[d]; sb += ey[d] * W[DD + d]; }
    alpha[r] = sa; beta[r] = sb;
}

// ---------- 3) U = Exb @ Eywb^T + alpha + beta, masked  (wave -> 16x64 tile) ----------
__global__ void cross_kernel(const bf16* __restrict__ Exb, const bf16* __restrict__ Eywb,
                             const float* __restrict__ alpha, const float* __restrict__ beta,
                             const unsigned char* __restrict__ mask_x,
                             const unsigned char* __restrict__ mask_y,
                             float* __restrict__ U) {
    int gw   = (blockIdx.x * blockDim.x + threadIdx.x) >> 5;  // global wave id
    int lane = threadIdx.x & 31;
    int ng = gw & 15;            // 16 column groups of 64
    int mt = (gw >> 4) & 63;     // 64 row tiles of 16
    int b  = gw >> 10;
    const bf16* Ea = Exb  + (size_t)b * LL * DD;
    const bf16* Eb = Eywb + (size_t)b * LL * DD;
    int row0 = mt * 16, col0 = ng * 64;
    int m = lane & 15, ksel = lane >> 4;

    v8f acc[4] = {};
    for (int k0 = 0; k0 < DD; k0 += 32) {
        v16bf a = load_frag16(Ea + (size_t)(row0 + m) * DD + k0 + ksel * 8);
#pragma unroll
        for (int nt = 0; nt < 4; ++nt) {
            // B operand is Eyw^T: column j of U is row j of Eyw -> K-contiguous
            v16bf bb = load_frag16(Eb + (size_t)(col0 + nt * 16 + m) * DD + k0 + ksel * 8);
            acc[nt] = __builtin_amdgcn_wmma_f32_16x16x32_bf16(
                false, a, false, bb, (short)0, acc[nt], false, false);
        }
    }
    int half = lane >> 4, n = lane & 15;
    float* Ub = U + (size_t)b * LL * LL;
#pragma unroll
    for (int nt = 0; nt < 4; ++nt)
#pragma unroll
        for (int r = 0; r < 8; ++r) {
            int i = row0 + half * 8 + r;
            int j = col0 + nt * 16 + n;
            float u = acc[nt][r] + alpha[b * LL + i] + beta[b * LL + j];
            bool valid = mask_x[b * LL + i] && mask_y[b * LL + j];
            Ub[(size_t)i * LL + j] = valid ? u : -1.0e9f;
        }
}

// ---------- 4a) A = softmax over axis=2 (rows of U) -> bf16 ----------
__global__ void softmax_row_kernel(const float* __restrict__ U, bf16* __restrict__ Abf) {
    int row = blockIdx.x;                       // b*LL + i ; row base is contiguous
    const float* u = U + (size_t)row * LL;
    __shared__ float red[256];
    int t = threadIdx.x;
    float mx = -3.4e38f;
    for (int j = t; j < LL; j += 256) mx = fmaxf(mx, u[j]);
    red[t] = mx; __syncthreads();
    for (int s = 128; s > 0; s >>= 1) { if (t < s) red[t] = fmaxf(red[t], red[t + s]); __syncthreads(); }
    mx = red[0]; __syncthreads();
    float sum = 0.f;
    for (int j = t; j < LL; j += 256) sum += __expf(u[j] - mx);
    red[t] = sum; __syncthreads();
    for (int s = 128; s > 0; s >>= 1) { if (t < s) red[t] += red[t + s]; __syncthreads(); }
    float inv = 1.0f / red[0];
    bf16* a = Abf + (size_t)row * LL;
    for (int j = t; j < LL; j += 256) a[j] = (bf16)(__expf(u[j] - mx) * inv);
}

// ---------- 4b) B[b,j,i] = softmax over axis=1 (columns of U) -> bf16 ----------
__global__ void softmax_col_kernel(const float* __restrict__ U, bf16* __restrict__ Bbf) {
    int cb = blockIdx.x; int b = cb >> 10; int j = cb & (LL - 1);
    const float* u = U + (size_t)b * LL * LL + j;
    __shared__ float red[256];
    int t = threadIdx.x;
    float mx = -3.4e38f;
    for (int i = t; i < LL; i += 256) mx = fmaxf(mx, u[(size_t)i * LL]);
    red[t] = mx; __syncthreads();
    for (int s = 128; s > 0; s >>= 1) { if (t < s) red[t] = fmaxf(red[t], red[t + s]); __syncthreads(); }
    mx = red[0]; __syncthreads();
    float sum = 0.f;
    for (int i = t; i < LL; i += 256) sum += __expf(u[(size_t)i * LL] - mx);
    red[t] = sum; __syncthreads();
    for (int s = 128; s > 0; s >>= 1) { if (t < s) red[t] += red[t + s]; __syncthreads(); }
    float inv = 1.0f / red[0];
    bf16* bo = Bbf + (size_t)b * LL * LL + (size_t)j * LL;
    for (int i = t; i < LL; i += 256) bo[i] = (bf16)(__expf(u[(size_t)i * LL] - mx) * inv);
}

// ---------- 5) generic WMMA GEMM: C[MxN] = A[MxK](bf16) * B[KxN](bf16) ----------
// block = 256 thr (8 waves) -> 128x64 of C.
// B tile (32x64) staged through LDS TRANSPOSED into fragment-major layout
// (row n, padded stride BPAD=40 elems -> conflict-free ds_load_b128 frag reads),
// with ping-pong double buffering (1 barrier / K-step, global load of tile k+1
// overlaps the 4 WMMAs of tile k).
// Writes f32 into Cf (ldcf may be 2560 = direct Gx/Gy column slot) and
// optionally a bf16 copy (for second-level GEMM operands).
__global__ void gemm_kernel(const bf16* __restrict__ Amat, size_t strideA, int lda,
                            const bf16* __restrict__ Bmat, size_t strideB, int ldb,
                            float* __restrict__ Cf, size_t strideCf, int ldcf,
                            bf16* __restrict__ Cb, size_t strideCb, int ldcb,
                            int Mtiles, int Ntiles, int K) {
    __shared__ __align__(16) bf16 Bls[2][64 * BPAD];
    int lane = threadIdx.x & 31, wave = threadIdx.x >> 5, tid = threadIdx.x;
    int job = blockIdx.x;
    int ng = job % Ntiles;
    int mt = (job / Ntiles) % Mtiles;
    int b  = job / (Ntiles * Mtiles);
    const bf16* Ab = Amat + (size_t)b * strideA;
    const bf16* Bb = Bmat + (size_t)b * strideB;
    int row0 = mt * 128 + wave * 16, col0 = ng * 64;
    int m = lane & 15, ksel = lane >> 4;

    // cooperative: coalesced global_load_b128 of row k, transpose-scatter into
    // fragment-major LDS layout Bl[n][k] (stride BPAD)
    int tr = tid >> 3;            // k within tile: 0..31
    int tc = (tid & 7) * 8;       // n0 within tile: 0..56
    auto load_tile = [&](int kk, int buf) {
        v8bf v = *(const v8bf*)(Bb + (size_t)(kk + tr) * ldb + col0 + tc);
        bf16* dst = &Bls[buf][0];
#pragma unroll
        for (int j = 0; j < 8; ++j) dst[(tc + j) * BPAD + tr] = v[j];
    };

    v8f acc[4] = {};
    load_tile(0, 0);
    int p = 0;
    for (int k0 = 0; k0 < K; k0 += 32) {
        __syncthreads();                              // tile p visible to all waves
        if (k0 + 32 < K) load_tile(k0 + 32, p ^ 1);   // prefetch next tile into other buffer

        const bf16* ap = Ab + (size_t)(row0 + m) * lda + k0 + ksel * 8;
        __builtin_prefetch(ap + 32, 0, 0);            // global_prefetch of next A chunk
        v16bf a = load_frag16(ap);
        const bf16* Bl = &Bls[p][0];
#pragma unroll
        for (int nt = 0; nt < 4; ++nt) {
            // two ds_load_b128 per fragment (K-contiguous in transposed tile)
            v16bf bfr = load_frag16(Bl + (nt * 16 + m) * BPAD + ksel * 8);
            acc[nt] = __builtin_amdgcn_wmma_f32_16x16x32_bf16(
                false, a, false, bfr, (short)0, acc[nt], false, false);
        }
        p ^= 1;
    }
    int half = lane >> 4, n = lane & 15;
    float* Cfb = Cf + (size_t)b * strideCf;
    bf16*  Cbb = Cb ? Cb + (size_t)b * strideCb : nullptr;
#pragma unroll
    for (int nt = 0; nt < 4; ++nt)
#pragma unroll
        for (int r = 0; r < 8; ++r) {
            int i = row0 + half * 8 + r;
            int j = col0 + nt * 16 + n;
            float v = acc[nt][r];
            Cfb[(size_t)i * ldcf + j] = v;
            if (Cbb) Cbb[(size_t)i * ldcb + j] = (bf16)v;
        }
}

// ---------- 6) fill Ex/Ey slots and Hadamard-product slots of Gx/Gy ----------
__global__ void finalize_kernel(const float* __restrict__ Ex, const float* __restrict__ Ey,
                                float* __restrict__ Gx, float* __restrict__ Gy, long n) {
    for (long idx = blockIdx.x * 256L + threadIdx.x; idx < n; idx += (long)gridDim.x * 256L) {
        long row = idx >> 9;            // b*LL + i
        int  c   = (int)(idx & (DD - 1));
        float ex = Ex[idx], ey = Ey[idx];
        float* gx = Gx + row * (5L * DD);
        float* gy = Gy + row * (5L * DD);
        float a1 = gx[DD + c],  a2 = gx[2 * DD + c];
        float b1 = gy[DD + c],  b2 = gy[2 * DD + c];
        gx[c] = ex; gx[3 * DD + c] = ex * a1; gx[4 * DD + c] = ex * a2;
        gy[c] = ey; gy[3 * DD + c] = ey * b1; gy[4 * DD + c] = ey * b2;
    }
}

extern "C" void kernel_launch(void* const* d_in, const int* in_sizes, int n_in,
                              void* d_out, int out_size, void* d_ws, size_t ws_size,
                              hipStream_t stream) {
    const float* Ex = (const float*)d_in[0];
    const float* Ey = (const float*)d_in[1];
    const unsigned char* mask_x = (const unsigned char*)d_in[2];
    const unsigned char* mask_y = (const unsigned char*)d_in[3];
    const float* W  = (const float*)d_in[4];

    const long nE = (long)BS * LL * DD;          // 4,194,304
    char* ws = (char*)d_ws;
    size_t off = 0;
    auto carve = [&](size_t bytes) -> void* {
        off = (off + 255) & ~(size_t)255;
        void* p = ws + off;
        off += bytes;
        return p;
    };
    bf16*  Exb   = (bf16*)carve(nE * 2);
    bf16*  Eyb   = (bf16*)carve(nE * 2);
    bf16*  Eywb  = (bf16*)carve(nE * 2);
    float* alpha = (float*)carve((size_t)BS * LL * 4);
    float* beta  = (float*)carve((size_t)BS * LL * 4);
    float* U     = (float*)carve((size_t)BS * LL * LL * 4);
    bf16*  Abf   = (bf16*)carve((size_t)BS * LL * LL * 2);
    bf16*  Bbf   = (bf16*)carve((size_t)BS * LL * LL * 2);
    bf16*  A1b   = (bf16*)carve(nE * 2);
    bf16*  B1b   = (bf16*)carve(nE * 2);

    float* Gx = (float*)d_out;                         // (BS, LL, 5*DD)
    float* Gy = Gx + (size_t)BS * LL * (5 * DD);

    prep_kernel<<<4096, 256, 0, stream>>>(Ex, Ey, W, Exb, Eyb, Eywb, nE);
    alphabeta_kernel<<<(BS * LL + 255) / 256, 256, 0, stream>>>(Ex, Ey, W, alpha, beta);

    // U: 8192 wave-jobs, 8 waves/block -> 1024 blocks
    cross_kernel<<<1024, 256, 0, stream>>>(Exb, Eywb, alpha, beta, mask_x, mask_y, U);

    softmax_row_kernel<<<BS * LL, 256, 0, stream>>>(U, Abf);
    softmax_col_kernel<<<BS * LL, 256, 0, stream>>>(U, Bbf);

    const size_t sAB = (size_t)LL * LL;   // operand batch stride (A/B matrices)
    const size_t sE  = (size_t)LL * DD;   // operand batch stride (d=512 matrices)
    const size_t sG  = (size_t)LL * 5 * DD;
    // A_1bar = A @ Ey   -> Gx[:, 512:1024], bf16 copy to A1b
    gemm_kernel<<<512, 256, 0, stream>>>(Abf, sAB, LL, Eyb, sE, DD,
                                         Gx + DD, sG, 5 * DD, A1b, sE, DD, 8, 8, LL);
    // B_1bar = B @ Ex   -> Gy[:, 512:1024], bf16 copy to B1b
    gemm_kernel<<<512, 256, 0, stream>>>(Bbf, sAB, LL, Exb, sE, DD,
                                         Gy + DD, sG, 5 * DD, B1b, sE, DD, 8, 8, LL);
    // A_2bar = A @ B_1bar -> Gx[:, 1024:1536]
    gemm_kernel<<<512, 256, 0, stream>>>(Abf, sAB, LL, B1b, sE, DD,
                                         Gx + 2 * DD, sG, 5 * DD, (bf16*)nullptr, 0, 0, 8, 8, LL);
    // B_2bar = B @ A_1bar -> Gy[:, 1024:1536]
    gemm_kernel<<<512, 256, 0, stream>>>(Bbf, sAB, LL, A1b, sE, DD,
                                         Gy + 2 * DD, sG, 5 * DD, (bf16*)nullptr, 0, 0, 8, 8, LL);

    finalize_kernel<<<4096, 256, 0, stream>>>(Ex, Ey, Gx, Gy, nE);
}
